// MSIN_2645699854541
// MI455X (gfx1250) — matrix operand order repaired and checked by reading, not compiled, not run on hardware
//
#include <hip/hip_runtime.h>
#include <hip/hip_bf16.h>

// Problem constants (from reference): B,T,D,K,H,A = 64,512,128,8,128,128
#define Bn 64
#define Tn 512
#define Dn 128
#define Kn 8
#define Hn 128
#define An 128
#define ZK 384          // 2D + H, LSTM input width
#define GN 512          // 4H, gate width

// Output layout (flat, fp32): outputs[B,T,H] | P[B,T,K] | h_f[B,H] | c_f[B,H]
#define OUT_P  (Bn*Tn*Hn)                 // 4194304
#define OUT_HF (OUT_P + Bn*Tn*Kn)         // 4456448
#define OUT_CF (OUT_HF + Bn*Hn)           // 4464640

// Workspace layout (bf16 repacked weights, element offsets in ushorts)
#define WS_WL 0                 // 384 frags * 512 halves
#define WS_WS 196608            // 32 frags * 512
#define WS_WH 212992            // 32 frags * 512  (end: 458752 bytes)

typedef __attribute__((ext_vector_type(16))) __bf16          v16bf;
typedef __attribute__((ext_vector_type(16))) unsigned short  v16us;
typedef __attribute__((ext_vector_type(8)))  float           v8f;

union BFV { v16us u; v16bf b; };

__device__ __forceinline__ unsigned short f2bf(float f) {
  union { float f; unsigned u; } c; c.f = f;
  unsigned u = c.u + 0x7FFFu + ((c.u >> 16) & 1u);   // round-to-nearest-even
  return (unsigned short)(u >> 16);
}

// Offset within one 512-half fragment for element (m in 0..15, k)
// A 16x32 bf16 layout: lane l -> m=l%16; element e -> K=(e/8)*16+(l/16)*8+e%8
__device__ __forceinline__ int packedIdx(int m, int k) {
  int lane = (((k & 15) >> 3) << 4) + m;          // ((k%16)/8)*16 + m
  int e    = (((k & 31) >> 4) << 3) + (k & 7);    // ((k%32)/16)*8 + k%8
  return lane * 16 + e;
}

// Any packed fragment (A or B): 32 B per lane, contiguous -> 2x b128 loads
__device__ __forceinline__ v16bf loadFrag(const unsigned short* p, int lane) {
  BFV r;
  r.u = *(const v16us*)(p + lane * 16);
  return r.b;
}

__device__ __forceinline__ v8f wmma_bf16(v16bf a, v16bf b, v8f c) {
  return __builtin_amdgcn_wmma_f32_16x16x32_bf16(
      false, a, false, b, (short)0, c, false, false);
}

// ---- Kernel 1: repack fp32 weights into bf16 B-fragment order --------------
// B 32x16: lane l -> n=l%16, K=(e/8)*16+(l/16)*8+e%8
__global__ void repack_weights(const float* __restrict__ Wl,
                               const float* __restrict__ Ws,
                               const float* __restrict__ Wh,
                               unsigned short* __restrict__ wsOut) {
  int f = blockIdx.x, lane = threadIdx.x;
  const float* src; int ld, kBase, nBase; unsigned short* dst;
  if (f < 384) {
    int nt = f / 12, kt = f % 12;
    src = Wl; ld = GN; kBase = kt * 32; nBase = nt * 16;
    dst = wsOut + WS_WL + f * 512;
  } else if (f < 416) {
    int g = f - 384; int nt = g / 4, kt = g % 4;
    src = Ws; ld = An; kBase = kt * 32; nBase = nt * 16;
    dst = wsOut + WS_WS + g * 512;
  } else {
    int g = f - 416; int nt = g / 4, kt = g % 4;
    src = Wh; ld = An; kBase = kt * 32; nBase = nt * 16;
    dst = wsOut + WS_WH + g * 512;
  }
  int n  = nBase + (lane & 15);
  int kb = kBase + ((lane >> 4) << 3);
#pragma unroll
  for (int e = 0; e < 16; ++e)
    dst[lane * 16 + e] = f2bf(src[(kb + ((e >> 3) << 4) + (e & 7)) * ld + n]);
}

// ---- Kernel 2: persistent recurrence, one WG per 16 batch rows -------------
// LDS carve (bytes)
#define SM_ST    0        // sT   [128][128] f32 (rows b*8+k, cols d; for ctx)
#define SM_EPRE  65536    // epre [128][128] f32 (s@Ws result)
#define SM_HWH   131072   // hwh  [16][128]  f32 (h@Wh result)
#define SM_ZBUF  139264   // zbuf [16][512]  f32 (gate pre-activations)
#define SM_H     172032   // hS   [16][128]  f32
#define SM_C     180224   // cS   [16][128]  f32
#define SM_PB    188416   // Pb   [16][8]
#define SM_EARR  188928   // eArr [128]
#define SM_RED   189440   // red  [128][4]
#define SM_BIAS  191488   // bias [512]
#define SM_VATT  193536   // vatt [128]
#define SM_WSF   194048   // W_s B-frags, 32 KB bf16
#define SM_WHF   226816   // W_h B-frags, 32 KB bf16
#define SM_AS    259584   // s-tile A-frags (8mt x 4kt), 32 KB bf16
#define SM_AZ    292352   // [x|ctx|h] A-frags (12 kt),  12 KB bf16
#define SM_AH    304640   // h A-frags (4 kt),            4 KB bf16
#define SM_TOTAL 308736

__global__ __launch_bounds__(512, 1) void rnn_kernel(
    const float* __restrict__ x, const float* __restrict__ s,
    const int* __restrict__ seqlen, const float* __restrict__ b_lstm,
    const float* __restrict__ v_att, const unsigned short* __restrict__ wsp,
    float* __restrict__ out) {
  extern __shared__ char smem[];
  float* sT   = (float*)(smem + SM_ST);
  float* epre = (float*)(smem + SM_EPRE);
  float* hwh  = (float*)(smem + SM_HWH);
  float* zbuf = (float*)(smem + SM_ZBUF);
  float* hS   = (float*)(smem + SM_H);
  float* cS   = (float*)(smem + SM_C);
  float* Pb   = (float*)(smem + SM_PB);
  float* eArr = (float*)(smem + SM_EARR);
  float* red  = (float*)(smem + SM_RED);
  float* bias = (float*)(smem + SM_BIAS);
  float* vat  = (float*)(smem + SM_VATT);
  unsigned short* wsF = (unsigned short*)(smem + SM_WSF);
  unsigned short* whF = (unsigned short*)(smem + SM_WHF);
  unsigned short* aS  = (unsigned short*)(smem + SM_AS);
  unsigned short* aZ  = (unsigned short*)(smem + SM_AZ);
  unsigned short* aH  = (unsigned short*)(smem + SM_AH);

  const int tid  = threadIdx.x;
  const int lane = tid & 31;
  const int wave = tid >> 5;
  const int bG0  = blockIdx.x * 16;
  const unsigned short* wlF = wsp + WS_WL;

  // ---- one-time init ----
  {
    const unsigned* srcS = (const unsigned*)(wsp + WS_WS);
    const unsigned* srcH = (const unsigned*)(wsp + WS_WH);
    unsigned* dS = (unsigned*)wsF; unsigned* dH = (unsigned*)whF;
#pragma unroll
    for (int i = 0; i < 16; ++i) {
      dS[tid + 512 * i] = srcS[tid + 512 * i];
      dH[tid + 512 * i] = srcH[tid + 512 * i];
    }
    bias[tid] = b_lstm[tid];
    if (tid < 128) vat[tid] = v_att[tid];
#pragma unroll
    for (int i = 0; i < 4; ++i) { hS[tid + 512 * i] = 0.f; cS[tid + 512 * i] = 0.f; }
    // zero packed h (bf16 of 0.0 == 0) in aZ h-part and aH
    unsigned* zZ = (unsigned*)aZ; unsigned* zH = (unsigned*)aH;
#pragma unroll
    for (int i = 0; i < 6; ++i) zZ[tid + 512 * i] = 0u;
    zH[tid + 512 * 0] = 0u; zH[tid + 512 * 1] = 0u;
  }
  __syncthreads();

  for (int t = 0; t < Tn; ++t) {
    // -- Phase 1: stage s_t (fp32 copy + packed bf16 A-frags) and x_t -------
    for (int i = 0; i < 32; ++i) {
      int flat = tid + 512 * i;            // 128 rows x 128 cols
      int m = flat >> 7, d = flat & 127;
      int b = m >> 3, k = m & 7;
      float g = s[(((bG0 + b) * Tn + t) * Kn + k) * Dn + d];
      sT[flat] = g;
      aS[(((m >> 4) << 2) + (d >> 5)) * 512 + packedIdx(m & 15, d)] = f2bf(g);
    }
#pragma unroll
    for (int i = 0; i < 4; ++i) {
      int flat = tid + 512 * i;
      int b = flat >> 7, d = flat & 127;
      float g = x[((bG0 + b) * Tn + t) * Dn + d];
      aZ[(d >> 5) * 512 + packedIdx(b, d)] = f2bf(g);   // z K-range [0,128)
    }
    if (t + 1 < Tn) {                      // prefetch next step's s-tile
      int flat = tid << 5;
      int m = flat >> 7, d = flat & 127;
      int b = m >> 3, k = m & 7;
      __builtin_prefetch(&s[(((bG0 + b) * Tn + (t + 1)) * Kn + k) * Dn + d], 0, 1);
    }
    __syncthreads();

    // -- Phase 2: epre = sT@W_s (128x128x128), hwh = h@W_h (16x128x128) -----
    // 4 SW tiles per wave, statically scheduled (no exec divergence).
#pragma unroll
    for (int it = 0; it < 4; ++it) {
      int q = wave + it * 16;              // 0..63
      int mt = q >> 3, nt = q & 7;
      v16bf afr[4], bfr[4];
#pragma unroll
      for (int kt = 0; kt < 4; ++kt) {
        afr[kt] = loadFrag(aS + (mt * 4 + kt) * 512, lane);
        bfr[kt] = loadFrag(wsF + (nt * 4 + kt) * 512, lane);
      }
      v8f acc = {};
#pragma unroll
      for (int kt = 0; kt < 4; ++kt) acc = wmma_bf16(afr[kt], bfr[kt], acc);
      int n  = nt * 16 + (lane & 15);
      int mB = mt * 16 + ((lane >> 4) << 3);
#pragma unroll
      for (int r = 0; r < 8; ++r) epre[(mB + r) * 128 + n] = acc[r];
    }
    if (wave >= 8) {                       // one hWh tile on waves 8..15
      int nt = wave - 8;
      v16bf afr[4], bfr[4];
#pragma unroll
      for (int kt = 0; kt < 4; ++kt) {
        afr[kt] = loadFrag(aH + kt * 512, lane);
        bfr[kt] = loadFrag(whF + (nt * 4 + kt) * 512, lane);
      }
      v8f acc = {};
#pragma unroll
      for (int kt = 0; kt < 4; ++kt) acc = wmma_bf16(afr[kt], bfr[kt], acc);
      int n  = nt * 16 + (lane & 15);
      int mB = (lane >> 4) << 3;
#pragma unroll
      for (int r = 0; r < 8; ++r) hwh[(mB + r) * 128 + n] = acc[r];
    }
    __syncthreads();

    // -- Phase 3: e = v . tanh(epre + hwh), softmax_K, ctx ------------------
    {
      int row = tid >> 2, part = tid & 3;
      const float* ep = epre + row * 128;
      const float* hp = hwh + (row >> 3) * 128;
      float sum = 0.f;
#pragma unroll
      for (int a2 = part * 32; a2 < part * 32 + 32; ++a2)
        sum += vat[a2] * tanhf(ep[a2] + hp[a2]);
      red[row * 4 + part] = sum;
    }
    __syncthreads();
    if (tid < 128)
      eArr[tid] = red[tid * 4] + red[tid * 4 + 1] + red[tid * 4 + 2] + red[tid * 4 + 3];
    __syncthreads();
    if (tid < 16) {
      int b = tid;
      float m = -1e30f;
#pragma unroll
      for (int k = 0; k < 8; ++k) m = fmaxf(m, eArr[b * 8 + k]);
      float ex[8], sm = 0.f;
#pragma unroll
      for (int k = 0; k < 8; ++k) { ex[k] = __expf(eArr[b * 8 + k] - m); sm += ex[k]; }
      float inv = 1.f / sm;
      int act = (t < seqlen[bG0 + b]) ? 1 : 0;
#pragma unroll
      for (int k = 0; k < 8; ++k) {
        float p = ex[k] * inv;
        Pb[b * 8 + k] = p;
        out[OUT_P + ((bG0 + b) * Tn + t) * Kn + k] = act ? p : 0.f;
      }
    }
    __syncthreads();
#pragma unroll
    for (int i = 0; i < 4; ++i) {          // ctx -> packed aZ (K-range [128,256))
      int flat = tid + 512 * i;
      int b = flat >> 7, d = flat & 127;
      float sum = 0.f;
#pragma unroll
      for (int k = 0; k < 8; ++k) sum += Pb[b * 8 + k] * sT[(b * 8 + k) * 128 + d];
      aZ[(4 + (d >> 5)) * 512 + packedIdx(b, d)] = f2bf(sum);
    }
    __syncthreads();

    // -- Phase 4: z = [x|ctx|h] @ W_lstm  (16 x 512 x 384) ------------------
    {
      v16bf af[12];
#pragma unroll
      for (int kt = 0; kt < 12; ++kt) af[kt] = loadFrag(aZ + kt * 512, lane);
#pragma unroll
      for (int j = 0; j < 2; ++j) {
        int nt = wave * 2 + j;
        v8f acc = {};
#pragma unroll
        for (int g = 0; g < 3; ++g) {      // B frags in groups of 4 (VGPR cap)
          v16bf bfr[4];
#pragma unroll
          for (int u = 0; u < 4; ++u)
            bfr[u] = loadFrag(wlF + (nt * 12 + g * 4 + u) * 512, lane);
#pragma unroll
          for (int u = 0; u < 4; ++u)
            acc = wmma_bf16(af[g * 4 + u], bfr[u], acc);
        }
        int n  = nt * 16 + (lane & 15);
        int mB = (lane >> 4) << 3;
#pragma unroll
        for (int r = 0; r < 8; ++r) zbuf[(mB + r) * GN + n] = acc[r] + bias[n];
      }
    }
    __syncthreads();

    // -- Phase 5: gates, state update, masked output, pack h for next step --
#pragma unroll
    for (int i = 0; i < 4; ++i) {
      int flat = tid + 512 * i;
      int b = flat >> 7, j = flat & 127;
      float zi = zbuf[b * GN + j];
      float zf = zbuf[b * GN + 128 + j];
      float zg = zbuf[b * GN + 256 + j];
      float zo = zbuf[b * GN + 384 + j];
      float co = cS[b * 128 + j];
      float fi = 1.f / (1.f + __expf(-zi));
      float ff = 1.f / (1.f + __expf(-zf));
      float fo = 1.f / (1.f + __expf(-zo));
      float cn = ff * co + fi * tanhf(zg);
      float hn = fo * tanhf(cn);
      bool act = t < seqlen[bG0 + b];
      float hv = act ? hn : hS[b * 128 + j];
      hS[b * 128 + j] = hv;
      cS[b * 128 + j] = act ? cn : co;
      out[((bG0 + b) * Tn + t) * Hn + j] = act ? hn : 0.f;
      unsigned short hb = f2bf(hv);
      int pi = packedIdx(b, j);
      aH[(j >> 5) * 512 + pi] = hb;              // for h@W_h next step
      aZ[(8 + (j >> 5)) * 512 + pi] = hb;        // z K-range [256,384)
    }
    __syncthreads();
  }

  // ---- finals: h_f, c_f ----
#pragma unroll
  for (int i = 0; i < 4; ++i) {
    int flat = tid + 512 * i;
    int b = flat >> 7, j = flat & 127;
    out[OUT_HF + (bG0 + b) * Hn + j] = hS[b * 128 + j];
    out[OUT_CF + (bG0 + b) * Hn + j] = cS[b * 128 + j];
  }
}

extern "C" void kernel_launch(void* const* d_in, const int* in_sizes, int n_in,
                              void* d_out, int out_size, void* d_ws, size_t ws_size,
                              hipStream_t stream) {
  const float* x    = (const float*)d_in[0];   // inputs [B,T,D]
  const float* s    = (const float*)d_in[1];   // s_inputs [B,T,K,D]
  const int*   seq  = (const int*)d_in[2];     // sequence_length [B]
  const float* Wl   = (const float*)d_in[3];   // W_lstm [384,512]
  const float* bl   = (const float*)d_in[4];   // b_lstm [512]
  const float* Ws   = (const float*)d_in[5];   // W_s [128,128]
  const float* Wh   = (const float*)d_in[6];   // W_h [128,128]
  const float* va   = (const float*)d_in[7];   // v_att [128]
  unsigned short* wsp = (unsigned short*)d_ws;
  float* out = (float*)d_out;

  repack_weights<<<448, 32, 0, stream>>>(Wl, Ws, Wh, wsp);
  rnn_kernel<<<4, 512, SM_TOTAL, stream>>>(x, s, seq, bl, va, wsp, out);
}
